// ConsolidationEngine_5007931867639
// MI455X (gfx1250) — compile-verified, matrix operand order checked by reading, not run on hardware
//
#include <hip/hip_runtime.h>

#define NITEMS  16384
#define DSTATE  2048
#define DSEM    1024
#define NSLOTS  4096
#define LRATE   0.01f
#define TH2     4.0f      // DIST_THRESH^2, compare squared distances
#define GBLOCKS 32        // 32 blocks * 8 waves = 256 waves = NSLOTS/16 slot-tiles
#define BATCH   64
#define NBATCH  (NITEMS / BATCH)
#define TILES_G ((BATCH / 16) * (BATCH / 16))    // 16 tiles of Gram

typedef __attribute__((ext_vector_type(16))) __bf16 v16bf;
typedef __attribute__((ext_vector_type(8)))  __bf16 v8bf;
typedef __attribute__((ext_vector_type(4)))  __bf16 v4bf;
typedef __attribute__((ext_vector_type(8)))  float  v8f;
typedef __attribute__((ext_vector_type(4)))  float  v4f;

// ---- load one 16x32 bf16 WMMA operand fragment (row-major, per-lane base) ----
__device__ __forceinline__ v16bf load_frag(const __bf16* p) {
  v8bf lo = *(const v8bf*)p;
  v8bf hi = *(const v8bf*)(p + 16);
  return __builtin_shufflevector(lo, hi, 0,1,2,3,4,5,6,7,8,9,10,11,12,13,14,15);
}

__device__ __forceinline__ v8f wmma_bf16(v16bf a, v16bf b, v8f c) {
  return __builtin_amdgcn_wmma_f32_16x16x32_bf16(
      /*neg_a=*/false, a, /*neg_b=*/false, b,
      /*c_mod=*/(short)0, c, /*reuse_a=*/false, /*reuse_b=*/false);
}

// ---------------- f32 -> bf16 conversion (vectorized) ----------------
__global__ void k_f32_to_bf16(const float* __restrict__ src,
                              __bf16* __restrict__ dst, int n4) {
  int i = blockIdx.x * blockDim.x + threadIdx.x;
  if (i >= n4) return;
  v4f v = ((const v4f*)src)[i];
  v4bf o;
  o.x = (__bf16)v.x; o.y = (__bf16)v.y; o.z = (__bf16)v.z; o.w = (__bf16)v.w;
  ((v4bf*)dst)[i] = o;
}

// ---------------- projection WMMA GEMM: C(M,N) = A(M,K) * B(N,K)^T + bias ----------------
// 2x2 register blocking: each wave computes a 32x32 output block (4 WMMA tiles),
// 8 fragment loads feed 4 WMMAs -> 32 B per lane of L2 traffic per WMMA.
__global__ void __launch_bounds__(256) k_gemm_wmma(
    const __bf16* __restrict__ A, const __bf16* __restrict__ B,
    const float* __restrict__ bias, float* __restrict__ C) {
  int gwave = (blockIdx.x * blockDim.x + threadIdx.x) >> 5;
  int lane  = threadIdx.x & 31;
  int sn = gwave & (DSEM / 32 - 1);          // 0..31  (N super-tile)
  int sm = gwave >> 5;                       // 0..511 (M super-tile)
  if (sm >= NITEMS / 32) return;
  int half = lane >> 4, lrow = lane & 15;

  const __bf16* a0 = A + (size_t)(sm * 32 + lrow) * DSTATE + half * 8;
  const __bf16* a1 = a0 + (size_t)16 * DSTATE;
  const __bf16* b0 = B + (size_t)(sn * 32 + lrow) * DSTATE + half * 8;
  const __bf16* b1 = b0 + (size_t)16 * DSTATE;

  v8f acc00 = {}, acc01 = {}, acc10 = {}, acc11 = {};
  for (int k = 0; k < DSTATE; k += 32) {
    v16bf av0 = load_frag(a0 + k);
    v16bf av1 = load_frag(a1 + k);
    v16bf bv0 = load_frag(b0 + k);
    v16bf bv1 = load_frag(b1 + k);
    acc00 = wmma_bf16(av0, bv0, acc00);
    acc01 = wmma_bf16(av0, bv1, acc01);
    acc10 = wmma_bf16(av1, bv0, acc10);
    acc11 = wmma_bf16(av1, bv1, acc11);
  }
  int col0 = sn * 32 + lrow, col1 = col0 + 16;
  int rowA = sm * 32 + half * 8, rowB = rowA + 16;
  float bias0 = bias[col0], bias1 = bias[col1];
#pragma unroll
  for (int r = 0; r < 8; ++r) {
    C[(size_t)(rowA + r) * DSEM + col0] = acc00[r] + bias0;
    C[(size_t)(rowA + r) * DSEM + col1] = acc01[r] + bias1;
    C[(size_t)(rowB + r) * DSEM + col0] = acc10[r] + bias0;
    C[(size_t)(rowB + r) * DSEM + col1] = acc11[r] + bias1;
  }
}

// ---------------- per-row squared norms of content ----------------
__device__ __forceinline__ float waveReduceSum(float v) {
#pragma unroll
  for (int off = 16; off > 0; off >>= 1) v += __shfl_xor(v, off, 32);
  return v;
}

__global__ void __launch_bounds__(256) k_rownorm(const float* __restrict__ X,
                                                 float* __restrict__ out) {
  __shared__ float ws[8];
  const v4f* row = (const v4f*)(X + (size_t)blockIdx.x * DSEM);
  v4f v = row[threadIdx.x];
  float s = v.x * v.x + v.y * v.y + v.z * v.z + v.w * v.w;
  s = waveReduceSum(s);
  int lane = threadIdx.x & 31, wave = threadIdx.x >> 5;
  if (lane == 0) ws[wave] = s;
  __syncthreads();
  if (threadIdx.x == 0) {
    float t = 0.f;
#pragma unroll
    for (int w = 0; w < 8; ++w) t += ws[w];
    out[blockIdx.x] = t;
  }
}

// ---------------- init: copy initial traces/strengths, zero state ----------------
__global__ void k_init(const float* __restrict__ tr_in, const float* __restrict__ st_in,
                       float* __restrict__ tr_out, float* __restrict__ st_out,
                       float* __restrict__ tnorm2, int* __restrict__ ctrl) {
  size_t i = (size_t)blockIdx.x * blockDim.x + threadIdx.x;
  size_t stride = (size_t)gridDim.x * blockDim.x;
  for (size_t j = i; j < (size_t)NSLOTS * DSEM; j += stride) tr_out[j] = tr_in[j];
  for (size_t j = i; j < NSLOTS; j += stride) { st_out[j] = st_in[j]; tnorm2[j] = 0.f; }
  if (i < 2) ctrl[i] = 0;
}

// ---------------- grid-wide barrier (persistent blocks, atomics + s_sleep) ----------------
__device__ __forceinline__ void grid_barrier(int* cnt, int* gen, int nblocks, int* mygen) {
  __syncthreads();
  if (threadIdx.x == 0) {
    int g = ++(*mygen);
    __threadfence();
    int prev = atomicAdd(cnt, 1);
    if (prev == nblocks - 1) {
      atomicExch(cnt, 0);
      __threadfence();
      atomicExch(gen, g);
    } else {
      while (((volatile int*)gen)[0] < g) __builtin_amdgcn_s_sleep(1);
    }
    __threadfence();
  }
  __syncthreads();
}

// ---------------- batched consolidation scan (persistent, GBLOCKS blocks) ----------------
// Per batch of 64 items:
//   phase 1 (all blocks, WMMA): D[64][4096] = traces_bf @ Cb^T ; Gram[64][64] = Cb @ Cb^T
//   phase 2 (block 0 only):     64 serial steps using D lookups + rank-1 D corrections
__global__ void __launch_bounds__(256, 1) k_consolidate(
    const float* __restrict__ content, const __bf16* __restrict__ content_bf,
    const float* __restrict__ cnorm2, const float* __restrict__ rewards,
    float* traces, __bf16* traces_bf, float* strengths, float* tnorm2,
    float* D, float* Gram, int* ctrl, float* scalars) {
  __shared__ float wdist[8];
  __shared__ int   widx[8];
  __shared__ int   s_do, s_slot, s_ptr, s_num;
  __shared__ float s_lr, s_nrm;

  const int tid = threadIdx.x, blk = blockIdx.x;
  const int lane = tid & 31, wave = tid >> 5;
  const int half = lane >> 4, lrow = lane & 15;
  int bar_gen = 0;

  if (blk == 0 && tid == 0) { s_ptr = 0; s_num = 0; }
  __syncthreads();

  for (int b = 0; b < NBATCH; ++b) {
    const int i0 = b * BATCH;
    const __bf16* Cb = content_bf + (size_t)i0 * DSEM;

    // ---- phase 1: cooperative WMMA GEMMs ----
    // D-GEMM, 1x4 blocking: wave gw owns slot-tile tm=gw and all 4 item-tiles;
    // the traces fragment is loaded once and feeds 4 WMMAs.
    int gw = blk * 8 + wave;                 // 0..255 == slot tile index
    {
      int tm = gw;
      const __bf16* ar  = traces_bf + (size_t)(tm * 16 + lrow) * DSEM + half * 8;
      const __bf16* br0 = Cb + (size_t)lrow * DSEM + half * 8;
      v8f acc[4] = {{}, {}, {}, {}};
      for (int k = 0; k < DSEM; k += 32) {
        v16bf av = load_frag(ar + k);
#pragma unroll
        for (int tn = 0; tn < 4; ++tn) {
          v16bf bv = load_frag(br0 + (size_t)tn * 16 * DSEM + k);
          acc[tn] = wmma_bf16(av, bv, acc[tn]);
        }
      }
      int s0 = tm * 16 + half * 8;
#pragma unroll
      for (int tn = 0; tn < 4; ++tn) {
        int jcol = tn * 16 + lrow;           // item index in batch
#pragma unroll
        for (int r = 0; r < 8; ++r)
          D[(size_t)jcol * NSLOTS + s0 + r] = acc[tn][r];   // item-major D[j][s]
      }
    }
    // Gram tiles on the first 16 waves
    if (gw < TILES_G) {
      int tm = gw >> 2, tn = gw & 3;
      const __bf16* ar = Cb + (size_t)(tm * 16 + lrow) * DSEM + half * 8;
      const __bf16* br = Cb + (size_t)(tn * 16 + lrow) * DSEM + half * 8;
      v8f acc = {};
      for (int k = 0; k < DSEM; k += 32)
        acc = wmma_bf16(load_frag(ar + k), load_frag(br + k), acc);
      int col = tn * 16 + lrow, row0 = tm * 16 + half * 8;
#pragma unroll
      for (int r = 0; r < 8; ++r)
        Gram[(row0 + r) * BATCH + col] = acc[r];            // Gram[t][j]
    }

    grid_barrier(&ctrl[0], &ctrl[1], GBLOCKS, &bar_gen);

    // ---- phase 2: serial scan over the batch, block 0 only ----
    if (blk == 0) {
      for (int j = 0; j < BATCH; ++j) {
        int item = i0 + j;
        int valid = s_num;
        float cn = cnorm2[item];
        const float* Dj = D + (size_t)j * NSLOTS;
        if (j + 1 < BATCH)
          __builtin_prefetch(D + (size_t)(j + 1) * NSLOTS + tid * 16, 0, 1);

        // each thread scans 16 contiguous slots (ascending -> strict < keeps lowest idx)
        float bestd = 3.0e38f; int bests = NSLOTS;
#pragma unroll
        for (int q = 0; q < 16; ++q) {
          int s = tid * 16 + q;
          if (s < valid) {
            float d2 = tnorm2[s] - 2.f * Dj[s] + cn;
            if (d2 < bestd) { bestd = d2; bests = s; }
          }
        }
#pragma unroll
        for (int off = 16; off > 0; off >>= 1) {
          float od = __shfl_xor(bestd, off, 32);
          int   oi = __shfl_xor(bests, off, 32);
          if (od < bestd || (od == bestd && oi < bests)) { bestd = od; bests = oi; }
        }
        if (lane == 0) { wdist[wave] = bestd; widx[wave] = bests; }
        __syncthreads();
        if (tid == 0) {
          float bd = 3.0e38f; int bi = NSLOTS;
#pragma unroll
          for (int w = 0; w < 8; ++w) {
            float d = wdist[w]; int ix = widx[w];
            if (d < bd || (d == bd && ix < bi)) { bd = d; bi = ix; }
          }
          int doup = (s_num > 0) && (bd < TH2);
          float rw = fabsf(rewards[item]);
          s_do = doup;
          if (doup) { s_slot = bi;    s_lr = LRATE * (1.f + rw); }
          else      { s_slot = s_ptr; s_lr = 1.0f; }
          s_nrm = 0.f;
        }
        __syncthreads();

        int slot = s_slot; float lr = s_lr; int doup = s_do;
        // row update: f32 traces + bf16 mirror, exact norm recompute
        v4f c = ((const v4f*)(content + (size_t)item * DSEM))[tid];
        v4f* trow = (v4f*)(traces + (size_t)slot * DSEM);
        v4f nt;
        if (doup) {
          v4f t = trow[tid];
          nt.x = t.x + lr * (c.x - t.x); nt.y = t.y + lr * (c.y - t.y);
          nt.z = t.z + lr * (c.z - t.z); nt.w = t.w + lr * (c.w - t.w);
        } else {
          nt = c;                        // exact insert
        }
        trow[tid] = nt;
        v4bf ntb;
        ntb.x = (__bf16)nt.x; ntb.y = (__bf16)nt.y;
        ntb.z = (__bf16)nt.z; ntb.w = (__bf16)nt.w;
        ((v4bf*)(traces_bf + (size_t)slot * DSEM))[tid] = ntb;

        float part = nt.x * nt.x + nt.y * nt.y + nt.z * nt.z + nt.w * nt.w;
        part = waveReduceSum(part);
        if (lane == 0) atomicAdd(&s_nrm, part);

        // rank-1 correction of D for the remaining items of this batch
        // (unified: insert has lr=1 -> D = Gram)
        if (tid < BATCH - 1 - j) {
          int j2 = j + 1 + tid;
          float g  = Gram[j * BATCH + j2];
          float dv = D[(size_t)j2 * NSLOTS + slot];
          D[(size_t)j2 * NSLOTS + slot] = (1.f - lr) * dv + lr * g;
        }
        __syncthreads();
        if (tid == 0) {
          tnorm2[slot] = s_nrm;
          if (doup) {
            strengths[slot] += 1.f;
          } else {
            strengths[slot] = 1.f;
            s_ptr = (s_ptr + 1) & (NSLOTS - 1);
            int n = s_num + 1; s_num = n > NSLOTS ? NSLOTS : n;
          }
        }
        __syncthreads();
      }
      if (tid == 0) __threadfence();   // publish traces_bf/D state before next batch
    }

    grid_barrier(&ctrl[0], &ctrl[1], GBLOCKS, &bar_gen);
  }

  // ---- final scalars (block 0) ----
  if (blk == 0) {
    int num = s_num;
    int valid = num > NSLOTS ? NSLOTS : num;
    float part = 0.f;
    for (int j = tid; j < valid; j += 256) part += strengths[j];
    part = waveReduceSum(part);
    if (lane == 0) wdist[wave] = part;
    __syncthreads();
    if (tid == 0) {
      float tot = 0.f;
#pragma unroll
      for (int w = 0; w < 8; ++w) tot += wdist[w];
      float denom = (float)(valid > 1 ? valid : 1);
      scalars[0] = (float)num;                       // num
      scalars[1] = (float)NITEMS;                    // num_consolidated
      scalars[2] = (num > 0) ? tot / denom : 0.f;    // mean_strength
    }
  }
}

// ---------------- host side ----------------
extern "C" void kernel_launch(void* const* d_in, const int* in_sizes, int n_in,
                              void* d_out, int out_size, void* d_ws, size_t ws_size,
                              hipStream_t stream) {
  const float* states    = (const float*)d_in[0];   // 16384 x 2048
  const float* rewards   = (const float*)d_in[1];   // 16384
  const float* W         = (const float*)d_in[2];   // 1024 x 2048
  const float* bias      = (const float*)d_in[3];   // 1024
  const float* traces_in = (const float*)d_in[4];   // 4096 x 1024
  const float* str_in    = (const float*)d_in[5];   // 4096

  char* ws = (char*)d_ws;
  size_t off = 0;
  auto alloc = [&](size_t bytes) {
    void* p = ws + off;
    off = (off + bytes + 255) & ~(size_t)255;
    return p;
  };
  __bf16* A_bf       = (__bf16*)alloc((size_t)NITEMS * DSTATE * 2);
  __bf16* W_bf       = (__bf16*)alloc((size_t)DSEM * DSTATE * 2);
  float*  content    = (float*) alloc((size_t)NITEMS * DSEM * 4);
  __bf16* content_bf = (__bf16*)alloc((size_t)NITEMS * DSEM * 2);
  __bf16* traces_bf  = (__bf16*)alloc((size_t)NSLOTS * DSEM * 2);
  float*  cnorm2     = (float*) alloc((size_t)NITEMS * 4);
  float*  tnorm2     = (float*) alloc((size_t)NSLOTS * 4);
  float*  D          = (float*) alloc((size_t)BATCH * NSLOTS * 4);
  float*  Gram       = (float*) alloc((size_t)BATCH * BATCH * 4);
  int*    ctrl       = (int*)   alloc(2 * 4);
  (void)ws_size; (void)in_sizes; (void)n_in; (void)out_size;

  float* traces_out = (float*)d_out;
  float* str_out    = traces_out + (size_t)NSLOTS * DSEM;
  float* scalars    = str_out + NSLOTS;

  // 1) bf16 staging
  { int n4 = NITEMS * DSTATE / 4; k_f32_to_bf16<<<n4 / 256, 256, 0, stream>>>(states, A_bf, n4); }
  { int n4 = DSEM * DSTATE / 4;   k_f32_to_bf16<<<n4 / 256, 256, 0, stream>>>(W, W_bf, n4); }
  { int n4 = NSLOTS * DSEM / 4;   k_f32_to_bf16<<<n4 / 256, 256, 0, stream>>>(traces_in, traces_bf, n4); }
  // 2) projection GEMM via WMMA (2x2 register blocking, 32x32 per wave)
  {
    int totalWaves = (NITEMS / 32) * (DSEM / 32);   // 16384
    k_gemm_wmma<<<totalWaves / 8, 256, 0, stream>>>(A_bf, W_bf, bias, content);
  }
  // 3) bf16 content + squared norms
  { int n4 = NITEMS * DSEM / 4;   k_f32_to_bf16<<<n4 / 256, 256, 0, stream>>>(content, content_bf, n4); }
  k_rownorm<<<NITEMS, 256, 0, stream>>>(content, cnorm2);
  // 4) initialize outputs + control state
  k_init<<<1024, 256, 0, stream>>>(traces_in, str_in, traces_out, str_out, tnorm2, ctrl);
  // 5) batched consolidation: WMMA distance GEMMs + serial corrections
  k_consolidate<<<GBLOCKS, 256, 0, stream>>>(content, content_bf, cnorm2, rewards,
                                             traces_out, str_out == nullptr ? nullptr : traces_bf,
                                             str_out, tnorm2, D, Gram, ctrl, scalars);
}
